// ConvLayer_50869592655509
// MI455X (gfx1250) — matrix operand order repaired and checked by reading, not compiled, not run on hardware
//
#include <hip/hip_runtime.h>
#include <hip/hip_bf16.h>
#include <stdint.h>

// ---------------------------------------------------------------------------
// Problem constants (from the reference)
// ---------------------------------------------------------------------------
static constexpr int NN  = 50000;     // nodes
static constexpr int EE  = 100000;    // edges
static constexpr int RE  = 200000;    // edge rep rows (2 per edge)
static constexpr int C5_ = 8000;
static constexpr int C6_ = 12000;
static constexpr int R5_ = 40000;     // 5 per C5 cycle
static constexpr int R6_ = 72000;     // 6 per C6 cycle
static constexpr int RC  = R5_ + R6_; // 112000 cycle rep rows
static constexpr int HC  = 128;      // hidden channels

typedef __attribute__((ext_vector_type(16))) __bf16 v16bf;
typedef __attribute__((ext_vector_type(8)))  float  v8f;

static __device__ __forceinline__ unsigned short bf16bits(float f) {
  __hip_bfloat16 h = __float2bfloat16(f);
  return *(unsigned short*)&h;
}

// ---------------------------------------------------------------------------
// Elementwise / data-movement kernels (all channel counts are multiples of 4,
// all column offsets multiples of 128 -> vectorize by float4 / uint2)
// ---------------------------------------------------------------------------
__global__ void k_zero(float* __restrict__ p, long n4) {
  long i = (long)blockIdx.x * 256 + threadIdx.x;
  if (i < n4) ((float4*)p)[i] = make_float4(0.f, 0.f, 0.f, 0.f);
}

__global__ void k_f32_to_bf16(const float* __restrict__ x, long n4,
                              __hip_bfloat16* __restrict__ y) {
  long i = (long)blockIdx.x * 256 + threadIdx.x;
  if (i >= n4) return;
  float4 v = ((const float4*)x)[i];
  union { unsigned short s[4]; uint2 u; } p;
  p.s[0] = bf16bits(v.x); p.s[1] = bf16bits(v.y);
  p.s[2] = bf16bits(v.z); p.s[3] = bf16bits(v.w);
  ((uint2*)y)[i] = p.u;
}

// out[r, co + 4*c4 ..] = tab[src(r), 4*c4 ..]   src(r) = idx ? idx[r] : r/div
__global__ void k_gather_f32(const float* __restrict__ tab,
                             const int* __restrict__ idx, int div,
                             long R, int C,
                             float* __restrict__ out, int ldo, int co) {
  const int C4 = C >> 2;
  long i = (long)blockIdx.x * 256 + threadIdx.x;
  if (i >= R * (long)C4) return;
  long r  = i / C4;
  int  c4 = (int)(i - r * C4);
  long sr = idx ? (long)idx[r] : (div > 1 ? r / div : r);
  float4 v = *((const float4*)(tab + sr * (long)C) + c4);
  *((float4*)(out + r * (long)ldo + co) + c4) = v;
}

__global__ void k_gather_bf16(const float* __restrict__ tab,
                              const int* __restrict__ idx, int div,
                              long R, int C,
                              __hip_bfloat16* __restrict__ out, int ldo, int co) {
  const int C4 = C >> 2;
  long i = (long)blockIdx.x * 256 + threadIdx.x;
  if (i >= R * (long)C4) return;
  long r  = i / C4;
  int  c4 = (int)(i - r * C4);
  long sr = idx ? (long)idx[r] : (div > 1 ? r / div : r);
  float4 v = *((const float4*)(tab + sr * (long)C) + c4);
  union { unsigned short s[4]; uint2 u; } p;
  p.s[0] = bf16bits(v.x); p.s[1] = bf16bits(v.y);
  p.s[2] = bf16bits(v.z); p.s[3] = bf16bits(v.w);
  *(uint2*)(out + r * (long)ldo + co + 4 * c4) = p.u;
}

// acc[idx[r], co + c] += x[(div>1 ? r/div : r), c]
__global__ void k_scatter_add(const float* __restrict__ x,
                              const int* __restrict__ idx, int div,
                              long R, int C,
                              float* __restrict__ acc, int lda, int co) {
  const int C4 = C >> 2;
  long i = (long)blockIdx.x * 256 + threadIdx.x;
  if (i >= R * (long)C4) return;
  long r  = i / C4;
  int  c4 = (int)(i - r * C4);
  float4 v = *((const float4*)(x + (div > 1 ? r / div : r) * (long)C) + c4);
  float* a = acc + (long)idx[r] * (long)lda + co + 4 * c4;
  atomicAdd(a + 0, v.x);
  atomicAdd(a + 1, v.y);
  atomicAdd(a + 2, v.z);
  atomicAdd(a + 3, v.w);
}

// s[inst, c] = sum_{k<g} x[inst*g + k, c]
__global__ void k_group_sum(const float* __restrict__ x, long nInst, int g, int C,
                            float* __restrict__ s) {
  const int C4 = C >> 2;
  long i = (long)blockIdx.x * 256 + threadIdx.x;
  if (i >= nInst * (long)C4) return;
  long inst = i / C4;
  int  c4   = (int)(i - inst * C4);
  const float4* p = (const float4*)(x + inst * (long)g * C) + c4;
  float ax = 0.f, ay = 0.f, az = 0.f, aw = 0.f;
  for (int k = 0; k < g; ++k) {
    float4 v = p[(long)k * C4];
    ax += v.x; ay += v.y; az += v.z; aw += v.w;
  }
  ((float4*)s)[i] = make_float4(ax, ay, az, aw);
}

// ---------------------------------------------------------------------------
// BF16 WMMA GEMM:  Y[M,Nc] = A[M,K] @ B[K,Nc]   (A,B bf16 row-major, Y f32)
// 256 threads = 8 waves; C tile 128x128; wave = 32x64 (2x4 WMMA 16x16x32).
// A tile staged via GLOBAL_LOAD_ASYNC_TO_LDS_B128 (ASYNCcnt), double-buffered.
// ---------------------------------------------------------------------------
#define BM 128
#define BN 128
#define BK 32

__global__ __launch_bounds__(256)
void k_gemm_bf16(const __hip_bfloat16* __restrict__ A,
                 const __hip_bfloat16* __restrict__ B,
                 float* __restrict__ Y,
                 int M, int K, int Nc) {
  __shared__ __align__(16) __hip_bfloat16 As[2][BM * BK]; // [row][k] row-major
  __shared__ __align__(16) __hip_bfloat16 Bs[2][BN * BK]; // [n][k]  transposed

  const int tid  = threadIdx.x;
  const int lane = tid & 31;
  const int wave = tid >> 5;
  const int wm   = wave & 3;                 // M offset 32*wm
  const int wn   = wave >> 2;                // N offset 64*wn
  const int rowBase = blockIdx.y * BM;
  const int colBase = blockIdx.x * BN;

  v8f acc[2][4];
#pragma unroll
  for (int i = 0; i < 2; ++i)
#pragma unroll
    for (int j = 0; j < 4; ++j)
#pragma unroll
      for (int e = 0; e < 8; ++e) acc[i][j][e] = 0.0f;

  const int arow = lane & 15;
  const int kg   = lane >> 4;   // K-half selector

  // ---- stagers -----------------------------------------------------------
  auto stageA_async = [&](int buf, int k0) {
    // 128x32 bf16 = 512 x 16B chunks; 2 per thread; async VMEM -> LDS
#pragma unroll
    for (int i = 0; i < 2; ++i) {
      int chunk = tid + i * 256;
      int r   = chunk >> 2;            // 4 chunks per row
      int seg = chunk & 3;
      int gr  = rowBase + r;
      if (gr >= M) gr = M - 1;         // clamp: OOB rows feed only discarded C
      const __hip_bfloat16* gp = A + (long)gr * K + k0 + seg * 8;
      unsigned lds = (unsigned)(unsigned long long)(uintptr_t)&As[buf][chunk * 8];
      asm volatile("global_load_async_to_lds_b128 %0, %1, off"
                   :: "v"(lds), "v"(gp) : "memory");
    }
  };
  auto stageB = [&](int buf, int k0) {
    // 32x128 bf16; load b128 along n, transpose-store b16 into [n][k]
    unsigned short* bs = (unsigned short*)&Bs[buf][0];
#pragma unroll
    for (int i = 0; i < 2; ++i) {
      int chunk = tid + i * 256;
      int kk = chunk >> 4;             // 16 chunks per k-row
      int np = (chunk & 15) * 8;       // starting n
      uint4 v = *(const uint4*)(B + (long)(k0 + kk) * Nc + colBase + np);
      unsigned sv[4] = {v.x, v.y, v.z, v.w};
#pragma unroll
      for (int e = 0; e < 4; ++e) {
        bs[(np + 2 * e + 0) * BK + kk] = (unsigned short)(sv[e] & 0xffffu);
        bs[(np + 2 * e + 1) * BK + kk] = (unsigned short)(sv[e] >> 16);
      }
    }
  };

  const int nT = K / BK;
  stageA_async(0, 0);
  stageB(0, 0);

  for (int t = 0; t < nT; ++t) {
    const int cur = t & 1;
    asm volatile("s_wait_asynccnt 0x0" ::: "memory"); // own async LDS writes done
    __syncthreads();                                   // tile published, prev reads done
    if (t + 1 < nT) {                                  // overlap next tile with WMMAs
      stageA_async(cur ^ 1, (t + 1) * BK);
      stageB(cur ^ 1, (t + 1) * BK);
    }

    // ---- per-lane fragments (ISA 16-bit A 16x32 / B 32x16 layouts) ----
    v16bf afrag[2];
#pragma unroll
    for (int tm = 0; tm < 2; ++tm) {
      int row = wm * 32 + tm * 16 + arow;
      const unsigned* ar = (const unsigned*)(&As[cur][0] + row * BK);
      union { unsigned u[8]; v16bf v; } ua;
#pragma unroll
      for (int v = 0; v < 8; ++v) {
        int kb = (v < 4) ? (kg * 8 + 2 * v) : (16 + kg * 8 + 2 * (v - 4));
        ua.u[v] = ar[kb >> 1];
      }
      afrag[tm] = ua.v;
    }
    v16bf bfrag[4];
#pragma unroll
    for (int tn = 0; tn < 4; ++tn) {
      int col = wn * 64 + tn * 16 + arow;
      const unsigned* br = (const unsigned*)(&Bs[cur][0] + col * BK);
      union { unsigned u[8]; v16bf v; } ub;
#pragma unroll
      for (int v = 0; v < 8; ++v) ub.u[v] = br[(v < 4) ? (kg * 4 + v) : (8 + kg * 4 + (v - 4))];
      bfrag[tn] = ub.v;
    }

#pragma unroll
    for (int tm = 0; tm < 2; ++tm)
#pragma unroll
      for (int tn = 0; tn < 4; ++tn)
        acc[tm][tn] = __builtin_amdgcn_wmma_f32_16x16x32_bf16(
            false, afrag[tm], false, bfrag[tn], (short)0, acc[tm][tn], false, false);
  }

  // ---- store C (f32 16x16: vgpr j -> M=j / j+8, N = lane%16) ----
#pragma unroll
  for (int tm = 0; tm < 2; ++tm) {
    int row0 = rowBase + wm * 32 + tm * 16 + (lane >> 4) * 8;
#pragma unroll
    for (int tn = 0; tn < 4; ++tn) {
      int col = colBase + wn * 64 + tn * 16 + (lane & 15);
#pragma unroll
      for (int j = 0; j < 8; ++j) {
        int r = row0 + j;
        if (r < M) Y[(long)r * Nc + col] = acc[tm][tn][j];
      }
    }
  }
}

// ---------------------------------------------------------------------------
// BatchNorm (training batch stats) + ReLU: stats / finalize / apply
// ---------------------------------------------------------------------------
__global__ void k_col_stats(const float* __restrict__ Y, int M, int C,
                            float* __restrict__ sums /* [C] sum, [C] sumsq */) {
  const int C4 = C >> 2;
  const int cg = threadIdx.x % C4;
  const int rw = threadIdx.x / C4;
  const int rowsPer = 256 / C4;     // 4 (C=256) or 8 (C=128)
  float s[4] = {0, 0, 0, 0}, q[4] = {0, 0, 0, 0};
  for (long r = (long)blockIdx.x * rowsPer + rw; r < M; r += (long)gridDim.x * rowsPer) {
    float4 v = *((const float4*)(Y + r * (long)C) + cg);
    float vv[4] = {v.x, v.y, v.z, v.w};
#pragma unroll
    for (int e = 0; e < 4; ++e) { s[e] += vv[e]; q[e] += vv[e] * vv[e]; }
  }
  __shared__ float sh[2048];
#pragma unroll
  for (int e = 0; e < 4; ++e) {
    sh[threadIdx.x * 4 + e]        = s[e];
    sh[1024 + threadIdx.x * 4 + e] = q[e];
  }
  __syncthreads();
  if (rw == 0) {
    for (int i = 1; i < rowsPer; ++i)
#pragma unroll
      for (int e = 0; e < 4; ++e) {
        s[e] += sh[(i * C4 + cg) * 4 + e];
        q[e] += sh[1024 + (i * C4 + cg) * 4 + e];
      }
#pragma unroll
    for (int e = 0; e < 4; ++e) {
      atomicAdd(&sums[4 * cg + e], s[e]);
      atomicAdd(&sums[C + 4 * cg + e], q[e]);
    }
  }
}

__global__ void k_bn_finalize(const float* __restrict__ sums,
                              const float* __restrict__ g, const float* __restrict__ b,
                              int C, float Minv, float* __restrict__ params) {
  int c = blockIdx.x * blockDim.x + threadIdx.x;
  if (c < C) {
    float mean = sums[c] * Minv;
    float var  = sums[C + c] * Minv - mean * mean;
    float sc   = g[c] * rsqrtf(var + 1e-5f);
    params[c]     = sc;
    params[C + c] = b[c] - mean * sc;
  }
}

__global__ void k_bn_apply(const float* __restrict__ Y, long total4, int C,
                           const float* __restrict__ params,
                           float* __restrict__ outF,
                           __hip_bfloat16* __restrict__ outB, int ldo, int co) {
  const int C4 = C >> 2;
  long i = (long)blockIdx.x * 256 + threadIdx.x;
  if (i >= total4) return;
  long r  = i / C4;
  int  c4 = (int)(i - r * C4);
  float4 v  = ((const float4*)Y)[i];
  float4 sc = ((const float4*)params)[c4];
  float4 sf = ((const float4*)(params + C))[c4];
  float o[4];
  o[0] = fmaxf(v.x * sc.x + sf.x, 0.f);
  o[1] = fmaxf(v.y * sc.y + sf.y, 0.f);
  o[2] = fmaxf(v.z * sc.z + sf.z, 0.f);
  o[3] = fmaxf(v.w * sc.w + sf.w, 0.f);
  if (outF) ((float4*)outF)[i] = make_float4(o[0], o[1], o[2], o[3]);
  if (outB) {
    union { unsigned short s[4]; uint2 u; } p;
    p.s[0] = bf16bits(o[0]); p.s[1] = bf16bits(o[1]);
    p.s[2] = bf16bits(o[2]); p.s[3] = bf16bits(o[3]);
    *(uint2*)(outB + r * (long)ldo + co + 4 * c4) = p.u;
  }
}

// ---------------------------------------------------------------------------
// Host orchestration
// ---------------------------------------------------------------------------
static inline long cdiv(long a, long b) { return (a + b - 1) / b; }

extern "C" void kernel_launch(void* const* d_in, const int* in_sizes, int n_in,
                              void* d_out, int out_size, void* d_ws, size_t ws_size,
                              hipStream_t stream) {
  (void)in_sizes; (void)n_in; (void)out_size; (void)ws_size;

  const float* node_rep  = (const float*)d_in[0];
  const float* edge_rep  = (const float*)d_in[1];
  const float* cycle_rep = (const float*)d_in[2];
  const int*   edge_nodes = (const int*)d_in[3];
  const int*   c5_nodes   = (const int*)d_in[4];
  const int*   c6_nodes   = (const int*)d_in[5];
  const float* W[9]  = { (const float*)d_in[6],  (const float*)d_in[9],
                         (const float*)d_in[12], (const float*)d_in[15],
                         (const float*)d_in[18], (const float*)d_in[21],
                         (const float*)d_in[24], (const float*)d_in[27],
                         (const float*)d_in[30] };
  const long Wn[9] = { 896L*256, 256L*128, 896L*256, 256L*128,
                       1408L*256, 256L*128, 640L*256, 256L*128, 256L*128 };
  const float* nm_g1=(const float*)d_in[7],  *nm_b1=(const float*)d_in[8];
  const float* nm_g2=(const float*)d_in[10], *nm_b2=(const float*)d_in[11];
  const float* em_g1=(const float*)d_in[13], *em_b1=(const float*)d_in[14];
  const float* em_g2=(const float*)d_in[16], *em_b2=(const float*)d_in[17];
  const float* ce_g1=(const float*)d_in[19], *ce_b1=(const float*)d_in[20];
  const float* ce_g2=(const float*)d_in[22], *ce_b2=(const float*)d_in[23];
  const float* cm_g1=(const float*)d_in[25], *cm_b1=(const float*)d_in[26];
  const float* cm_g2=(const float*)d_in[28], *cm_b2=(const float*)d_in[29];
  const float* fm_g =(const float*)d_in[31], *fm_b =(const float*)d_in[32];

  float* node_out  = (float*)d_out;
  float* edge_out  = node_out + (long)NN * HC;
  float* cycle_out = edge_out + (long)RE * HC;

  char* ws = (char*)d_ws;
  size_t off = 0;
  auto alloc = [&](size_t bytes) -> void* {
    off = (off + 255) & ~(size_t)255;
    void* p = ws + off;
    off += bytes;
    return p;
  };
  auto allocF = [&](long n) { return (float*)alloc((size_t)n * 4); };
  auto allocB = [&](long n) { return (__hip_bfloat16*)alloc((size_t)n * 2); };

  const int H = HC, H2 = 2*HC, H4 = 4*HC, H5 = 5*HC, H6 = 6*HC, H7 = 7*HC, H11 = 11*HC;

  auto zero = [&](float* p, long n) {
    k_zero<<<dim3((unsigned)cdiv(n/4,256)),256,0,stream>>>(p, n/4);
  };
  auto gatherF = [&](const float* tab, const int* idx, int div, long R, int C,
                     float* out, int ldo, int co) {
    k_gather_f32<<<dim3((unsigned)cdiv(R*(long)(C/4),256)),256,0,stream>>>(tab,idx,div,R,C,out,ldo,co);
  };
  auto gatherB = [&](const float* tab, const int* idx, int div, long R, int C,
                     __hip_bfloat16* out, int ldo, int co) {
    k_gather_bf16<<<dim3((unsigned)cdiv(R*(long)(C/4),256)),256,0,stream>>>(tab,idx,div,R,C,out,ldo,co);
  };
  auto scatter = [&](const float* x, const int* idx, int div, long R, int C,
                     float* acc, int lda, int co) {
    k_scatter_add<<<dim3((unsigned)cdiv(R*(long)(C/4),256)),256,0,stream>>>(x,idx,div,R,C,acc,lda,co);
  };
  auto gsum = [&](const float* x, long nI, int g, int C, float* s) {
    k_group_sum<<<dim3((unsigned)cdiv(nI*(long)(C/4),256)),256,0,stream>>>(x,nI,g,C,s);
  };

  __hip_bfloat16* Wb[9];
  for (int i = 0; i < 9; ++i) {
    Wb[i] = allocB(Wn[i]);
    k_f32_to_bf16<<<dim3((unsigned)cdiv(Wn[i]/4,256)),256,0,stream>>>(W[i], Wn[i]/4, Wb[i]);
  }

  float* stats  = allocF(512);
  float* params = allocF(512);

  auto gemm = [&](const __hip_bfloat16* A, const __hip_bfloat16* B, float* Y,
                  long M, int K, int Nc) {
    dim3 g((unsigned)(Nc / BN), (unsigned)cdiv(M, BM));
    k_gemm_bf16<<<g, 256, 0, stream>>>(A, B, Y, (int)M, K, Nc);
  };
  auto bn = [&](float* Y, long M, int C, const float* gam, const float* bet,
                float* outF, __hip_bfloat16* outB, int ldo, int co) {
    zero(stats, 2L * C);
    k_col_stats<<<dim3(1024),256,0,stream>>>(Y, (int)M, C, stats);
    k_bn_finalize<<<dim3(1),dim3(C),0,stream>>>(stats, gam, bet, C, 1.0f/(float)M, params);
    long total4 = M * (long)C / 4;
    k_bn_apply<<<dim3((unsigned)cdiv(total4,256)),256,0,stream>>>(Y, total4, C, params, outF, outB, ldo, co);
  };

  float*          Y1 = allocF((long)RE * 256);
  __hip_bfloat16* Hb = allocB((long)RE * 256);
  float*          Y2 = allocF((long)RE * 128);

  auto mlp2 = [&](const __hip_bfloat16* X, long M, int K,
                  const __hip_bfloat16* W1, const float* g1, const float* b1,
                  const __hip_bfloat16* W2, const float* g2, const float* b2,
                  float* outF, __hip_bfloat16* outB, int ldo, int co) {
    gemm(X, W1, Y1, M, K, 256);
    bn(Y1, M, 256, g1, b1, nullptr, Hb, 256, 0);
    gemm(Hb, W2, Y2, M, 256, 128);
    bn(Y2, M, 128, g2, b2, outF, outB, ldo, co);
  };

  // =========================================================================
  // Stage A: Edge_node
  // =========================================================================
  float* n2e0 = allocF((long)RE * H);
  gatherF(node_rep, edge_nodes, 1, RE, H, n2e0, H, 0);

  float* accA = allocF((long)NN * H); zero(accA, (long)NN * H);
  scatter(n2e0, edge_nodes, 1, RE, H, accA, H, 0);
  float* sA = allocF((long)EE * H);
  gsum(n2e0, EE, 2, H, sA);
  float* accB = allocF((long)NN * H); zero(accB, (long)NN * H);
  scatter(sA, edge_nodes, 2, RE, H, accB, H, 0);
  float* n2e = allocF((long)RE * H2);
  gatherF(accA, edge_nodes, 1, RE, H, n2e, H2, 0);
  gatherF(accB, edge_nodes, 1, RE, H, n2e, H2, H);

  float* accC = allocF((long)NN * H); zero(accC, (long)NN * H);
  scatter(edge_rep, edge_nodes, 1, RE, H, accC, H, 0);
  float* sC = allocF((long)EE * H);
  gsum(edge_rep, EE, 2, H, sC);
  float* accD = allocF((long)NN * H); zero(accD, (long)NN * H);
  scatter(sC, edge_nodes, 2, RE, H, accD, H, 0);
  float* e2e1 = allocF((long)RE * H2);
  gatherF(accC, edge_nodes, 1, RE, H, e2e1, H2, 0);
  gatherF(accD, edge_nodes, 1, RE, H, e2e1, H2, H);

  float* accE = allocF((long)NN * H2); zero(accE, (long)NN * H2);
  scatter(e2e1, edge_nodes, 1, RE, H2, accE, H2, 0);
  float* sE = allocF((long)EE * H2);
  gsum(e2e1, EE, 2, H2, sE);
  float* accF = allocF((long)NN * H2); zero(accF, (long)NN * H2);
  scatter(sE, edge_nodes, 2, RE, H2, accF, H2, 0);
  float* e2e = allocF((long)RE * H4);
  gatherF(accE, edge_nodes, 1, RE, H2, e2e, H4, 0);
  gatherF(accF, edge_nodes, 1, RE, H2, e2e, H4, H2);

  float* accG = allocF((long)NN * H6); zero(accG, (long)NN * H6);
  scatter(e2e, edge_nodes, 1, RE, H4, accG, H6, 0);
  scatter(n2e, edge_nodes, 1, RE, H2, accG, H6, H4);

  __hip_bfloat16* Xe1 = allocB((long)RE * H7);
  gatherB(edge_rep, nullptr, 1, RE, H,  Xe1, H7, 0);
  gatherB(e2e,      nullptr, 1, RE, H4, Xe1, H7, H);
  gatherB(n2e,      nullptr, 1, RE, H2, Xe1, H7, 5*H);
  __hip_bfloat16* Xn1 = allocB((long)NN * H7);
  gatherB(node_rep, nullptr, 1, NN, H,  Xn1, H7, 0);
  gatherB(accG,     nullptr, 1, NN, H6, Xn1, H7, H);

  __hip_bfloat16* Xfm = allocB((long)RE * H2);

  mlp2(Xe1, RE, H7, Wb[2], em_g1, em_b1, Wb[3], em_g2, em_b2,
       nullptr, Xfm, H2, 0);
  mlp2(Xn1, NN, H7, Wb[0], nm_g1, nm_b1, Wb[1], nm_g2, nm_b2,
       node_out, nullptr, 0, 0);

  // =========================================================================
  // Stage B: Edge_Cycle (reuses accC/accD)
  // =========================================================================
  float* e2c5a = allocF((long)R5_ * H2);
  gatherF(accC, c5_nodes, 1, R5_, H, e2c5a, H2, 0);
  gatherF(accD, c5_nodes, 1, R5_, H, e2c5a, H2, H);
  float* accH = allocF((long)NN * H2); zero(accH, (long)NN * H2);
  scatter(e2c5a, c5_nodes, 1, R5_, H2, accH, H2, 0);
  float* sH = allocF((long)C5_ * H2);
  gsum(e2c5a, C5_, 5, H2, sH);
  float* accI = allocF((long)NN * H2); zero(accI, (long)NN * H2);
  scatter(sH, c5_nodes, 5, R5_, H2, accI, H2, 0);
  float* e2c5 = allocF((long)R5_ * H4);
  gatherF(accH, c5_nodes, 1, R5_, H2, e2c5, H4, 0);
  gatherF(accI, c5_nodes, 1, R5_, H2, e2c5, H4, H2);

  float* e2c6a = allocF((long)R6_ * H2);
  gatherF(accC, c6_nodes, 1, R6_, H, e2c6a, H2, 0);
  gatherF(accD, c6_nodes, 1, R6_, H, e2c6a, H2, H);
  float* accJ = allocF((long)NN * H2); zero(accJ, (long)NN * H2);
  scatter(e2c6a, c6_nodes, 1, R6_, H2, accJ, H2, 0);
  float* sJ = allocF((long)C6_ * H2);
  gsum(e2c6a, C6_, 6, H2, sJ);
  float* accK = allocF((long)NN * H2); zero(accK, (long)NN * H2);
  scatter(sJ, c6_nodes, 6, R6_, H2, accK, H2, 0);
  float* e2c6 = allocF((long)R6_ * H4);
  gatherF(accJ, c6_nodes, 1, R6_, H2, e2c6, H4, 0);
  gatherF(accK, c6_nodes, 1, R6_, H2, e2c6, H4, H2);

  float* cycle_new = allocF((long)RC * H5);
  gatherF(e2c5, nullptr, 1, R5_, H4, cycle_new, H5, 0);
  gatherF(e2c6, nullptr, 1, R6_, H4, cycle_new + (long)R5_ * H5, H5, 0);
  gatherF(cycle_rep, nullptr, 1, RC, H, cycle_new, H5, H4);

  float* accL = allocF((long)NN * H5); zero(accL, (long)NN * H5);
  scatter(cycle_new, c5_nodes, 1, R5_, H5, accL, H5, 0);
  scatter(cycle_new + (long)R5_ * H5, c6_nodes, 1, R6_, H5, accL, H5, 0);
  float* sM = allocF((long)(C5_ + C6_) * H5);
  gsum(cycle_new, C5_, 5, H5, sM);
  gsum(cycle_new + (long)R5_ * H5, C6_, 6, H5, sM + (long)C5_ * H5);
  float* accM = allocF((long)NN * H5); zero(accM, (long)NN * H5);
  scatter(sM, c5_nodes, 5, R5_, H5, accM, H5, 0);
  scatter(sM + (long)C5_ * H5, c6_nodes, 6, R6_, H5, accM, H5, 0);

  __hip_bfloat16* Xce = allocB((long)RE * H11);
  gatherB(edge_rep, nullptr, 1, RE, H, Xce, H11, 0);
  gatherB(accL, edge_nodes, 1, RE, H5, Xce, H11, H);
  gatherB(accM, edge_nodes, 1, RE, H5, Xce, H11, 6*H);
  __hip_bfloat16* Xcm = allocB((long)RC * H5);
  gatherB(cycle_new, nullptr, 1, RC, H5, Xcm, H5, 0);

  mlp2(Xce, RE, H11, Wb[4], ce_g1, ce_b1, Wb[5], ce_g2, ce_b2,
       nullptr, Xfm, H2, H);
  mlp2(Xcm, RC, H5,  Wb[6], cm_g1, cm_b1, Wb[7], cm_g2, cm_b2,
       cycle_out, nullptr, 0, 0);

  gemm(Xfm, Wb[8], Y2, RE, 256, 128);
  bn(Y2, RE, 128, fm_g, fm_b, edge_out, nullptr, 0, 0);
}
